// RankingLossFunc_18545668784711
// MI455X (gfx1250) — compile-verified, matrix operand order checked by reading, not run on hardware
//
#include <hip/hip_runtime.h>

typedef __attribute__((ext_vector_type(16))) _Float16 v16h;
typedef __attribute__((ext_vector_type(8)))  _Float16 v8h;
typedef __attribute__((ext_vector_type(8)))  float    v8f;
typedef __attribute__((ext_vector_type(4)))  int      v4i;

#define DELTA 0.2f
#define EPS   1e-8f
#define D_DIM 1024
#define KSTEP 32
#define LDS_ROW 40   // 32 halves + 8 halves pad = 80B row stride (bank spread)

#define AS_GLOBAL __attribute__((address_space(1)))
#define AS_LDS    __attribute__((address_space(3)))

#if defined(__HIP_DEVICE_COMPILE__) && __has_builtin(__builtin_amdgcn_global_load_async_to_lds_b128)
#define HAVE_ASYNC_LDS 1
#else
#define HAVE_ASYNC_LDS 0
#endif

// ---------------- Pass 1: normalize q,k rows; pos = cos(q,k); qh = f16(qn) --
__global__ void __launch_bounds__(256)
norm_qk_kernel(const float* __restrict__ q, const float* __restrict__ k,
               _Float16* __restrict__ qh, float* __restrict__ pos) {
  const int row = blockIdx.x;
  const int tid = threadIdx.x;
  const float* qr = q + (size_t)row * D_DIM;
  const float* kr = k + (size_t)row * D_DIM;

  float qv[4], kv[4];
  float sq = 0.f, sk = 0.f, dq = 0.f;
#pragma unroll
  for (int j = 0; j < 4; ++j) {
    int i = tid + j * 256;
    qv[j] = qr[i]; kv[j] = kr[i];
    sq += qv[j] * qv[j];
    sk += kv[j] * kv[j];
    dq += qv[j] * kv[j];
  }
#pragma unroll
  for (int off = 16; off > 0; off >>= 1) {
    sq += __shfl_xor(sq, off, 32);
    sk += __shfl_xor(sk, off, 32);
    dq += __shfl_xor(dq, off, 32);
  }
  __shared__ float s_sq[8], s_sk[8], s_dq[8];
  const int w = tid >> 5, lane = tid & 31;
  if (lane == 0) { s_sq[w] = sq; s_sk[w] = sk; s_dq[w] = dq; }
  __syncthreads();
  float tsq = 0.f, tsk = 0.f, tdq = 0.f;
#pragma unroll
  for (int j = 0; j < 8; ++j) { tsq += s_sq[j]; tsk += s_sk[j]; tdq += s_dq[j]; }
  float nq = sqrtf(tsq); nq = nq > EPS ? nq : EPS;
  float nk = sqrtf(tsk); nk = nk > EPS ? nk : EPS;
  const float inq = 1.f / nq;
  if (tid == 0) pos[row] = tdq * inq * (1.f / nk);
  _Float16* out = qh + (size_t)row * D_DIM;
#pragma unroll
  for (int j = 0; j < 4; ++j) {
    int i = tid + j * 256;
    out[i] = (_Float16)(qv[j] * inq);
  }
}

// ---------------- Pass 2: normalize queue rows -> f16 ----------------------
__global__ void __launch_bounds__(256)
norm_queue_kernel(const float* __restrict__ queue, _Float16* __restrict__ qh) {
  const int row = blockIdx.x;
  const int tid = threadIdx.x;
  const float* qr = queue + (size_t)row * D_DIM;

  float qv[4];
  float sq = 0.f;
#pragma unroll
  for (int j = 0; j < 4; ++j) {
    int i = tid + j * 256;
    qv[j] = qr[i];
    sq += qv[j] * qv[j];
  }
#pragma unroll
  for (int off = 16; off > 0; off >>= 1) sq += __shfl_xor(sq, off, 32);
  __shared__ float s_sq[8];
  const int w = tid >> 5, lane = tid & 31;
  if (lane == 0) s_sq[w] = sq;
  __syncthreads();
  float tsq = 0.f;
#pragma unroll
  for (int j = 0; j < 8; ++j) tsq += s_sq[j];
  float nq = sqrtf(tsq); nq = nq > EPS ? nq : EPS;
  const float inq = 1.f / nq;
  _Float16* out = qh + (size_t)row * D_DIM;
#pragma unroll
  for (int j = 0; j < 4; ++j) {
    int i = tid + j * 256;
    out[i] = (_Float16)(qv[j] * inq);
  }
}

// --- Stage one 128x32 A tile + 128x32 B tile into LDS (async if available) --
// 1024 chunks of 16B total; 256 threads x 4 chunks (2 for A, 2 for B).
__device__ __forceinline__ void stage_tiles(const _Float16* __restrict__ gA,
                                            const _Float16* __restrict__ gB,
                                            _Float16* sAbuf, _Float16* sBbuf,
                                            int tid, int kb) {
#pragma unroll
  for (int j = 0; j < 2; ++j) {
    const int c   = tid + j * 256;     // chunk id 0..511
    const int row = c >> 2;            // 0..127
    const int col = (c & 3) * 8;       // halves within K-step
    const _Float16* ga = gA + (size_t)row * D_DIM + kb + col;
    const _Float16* gb = gB + (size_t)row * D_DIM + kb + col;
    _Float16* la = sAbuf + row * LDS_ROW + col;
    _Float16* lb = sBbuf + row * LDS_ROW + col;
#if HAVE_ASYNC_LDS
    __builtin_amdgcn_global_load_async_to_lds_b128(
        (AS_GLOBAL v4i*)ga, (AS_LDS v4i*)la, 0, 0);
    __builtin_amdgcn_global_load_async_to_lds_b128(
        (AS_GLOBAL v4i*)gb, (AS_LDS v4i*)lb, 0, 0);
#else
    *(v8h*)la = *(const v8h*)ga;
    *(v8h*)lb = *(const v8h*)gb;
#endif
  }
}

__device__ __forceinline__ void wait_async_grp_prev() {
#if HAVE_ASYNC_LDS
#if __has_builtin(__builtin_amdgcn_s_wait_asynccnt)
  __builtin_amdgcn_s_wait_asynccnt(4);
#else
  asm volatile("s_wait_asynccnt 0x4" ::: "memory");
#endif
#endif
}
__device__ __forceinline__ void wait_async_all() {
#if HAVE_ASYNC_LDS
#if __has_builtin(__builtin_amdgcn_s_wait_asynccnt)
  __builtin_amdgcn_s_wait_asynccnt(0);
#else
  asm volatile("s_wait_asynccnt 0x0" ::: "memory");
#endif
#endif
}

// ------- Pass 3: LDS-staged WMMA GEMM (qn @ queuen^T) + fused hinge --------
// Block = 256 threads = 8 waves; block tile 128(M) x 128(N), K-step 32,
// double-buffered LDS fed by async global->LDS copies (ASYNCcnt).
// Wave grid 2(M) x 4(N): each wave owns 64x32 = 4x2 tiles of 16x16.
__global__ void __launch_bounds__(256)
hinge_gemm_kernel(const _Float16* __restrict__ qh,
                  const _Float16* __restrict__ queueh,
                  const float* __restrict__ pos,
                  float* __restrict__ partials) {
  __shared__ _Float16 sA[2][128 * LDS_ROW];
  __shared__ _Float16 sB[2][128 * LDS_ROW];

  const int tid    = threadIdx.x;
  const int waveId = tid >> 5;
  const int lane   = tid & 31;
  const int laneLo = lane & 15;
  const int laneHi = lane >> 4;

  const int wm = waveId & 1;                  // 0..1  (M direction)
  const int wn = waveId >> 1;                 // 0..3  (N direction)
  const int m0 = blockIdx.y * 128 + wm * 64;  // wave M base (global)
  const int n0 = blockIdx.x * 128 + wn * 32;  // wave N base (global)
  (void)n0;

  const _Float16* gA = qh     + (size_t)blockIdx.y * 128 * D_DIM;
  const _Float16* gB = queueh + (size_t)blockIdx.x * 128 * D_DIM;

  v8f acc[4][2] = {};

  // LDS fragment offsets (ISA 7.12.2 16-bit A/B layout): lane l&15 holds
  // row l&15; halves map to two contiguous 16B chunks at +16B for l>=16.
  int aOff[4], bOff[2];
#pragma unroll
  for (int tm = 0; tm < 4; ++tm)
    aOff[tm] = (wm * 64 + tm * 16 + laneLo) * LDS_ROW + 8 * laneHi;
#pragma unroll
  for (int tn = 0; tn < 2; ++tn)
    bOff[tn] = (wn * 32 + tn * 16 + laneLo) * LDS_ROW + 8 * laneHi;

  // Prologue: stage K-step 0 into buffer 0.
  stage_tiles(gA, gB, &sA[0][0], &sB[0][0], tid, 0);

  const int NK = D_DIM / KSTEP;
  for (int i = 0; i < NK; ++i) {
    const int kb  = i * KSTEP;
    const int cur = i & 1;

    if (i + 1 < NK) {
      stage_tiles(gA, gB, &sA[cur ^ 1][0], &sB[cur ^ 1][0], tid, kb + KSTEP);
      wait_async_grp_prev();   // asynccnt <= 4: current buffer's copies done
    } else {
      wait_async_all();
    }
    __syncthreads();           // whole block's staging of buf[cur] visible

    const _Float16* sAc = &sA[cur][0];
    const _Float16* sBc = &sB[cur][0];
    v16h aF[4], bF[2];
#pragma unroll
    for (int tm = 0; tm < 4; ++tm) {
      v8h lo = *(const v8h*)(sAc + aOff[tm]);
      v8h hi = *(const v8h*)(sAc + aOff[tm] + 16);
      aF[tm] = __builtin_shufflevector(lo, hi, 0,1,2,3,4,5,6,7,8,9,10,11,12,13,14,15);
    }
#pragma unroll
    for (int tn = 0; tn < 2; ++tn) {
      v8h lo = *(const v8h*)(sBc + bOff[tn]);
      v8h hi = *(const v8h*)(sBc + bOff[tn] + 16);
      bF[tn] = __builtin_shufflevector(lo, hi, 0,1,2,3,4,5,6,7,8,9,10,11,12,13,14,15);
    }
#pragma unroll
    for (int tm = 0; tm < 4; ++tm)
#pragma unroll
      for (int tn = 0; tn < 2; ++tn)
        acc[tm][tn] = __builtin_amdgcn_wmma_f32_16x16x32_f16(
            false, aF[tm], false, bF[tn], (short)0, acc[tm][tn], false, false);

    __syncthreads();           // all reads of buf[cur] done before re-staging
  }

  // Fused hinge epilogue on accumulator registers.
  // C/D layout: VGPR r, lanes 0-15 -> M=r, lanes 16-31 -> M=r+8; N=lane&15.
  float lsum = 0.f;
#pragma unroll
  for (int tm = 0; tm < 4; ++tm) {
#pragma unroll
    for (int r = 0; r < 8; ++r) {
      const float dm = DELTA - pos[m0 + tm * 16 + r + 8 * laneHi];
#pragma unroll
      for (int tn = 0; tn < 2; ++tn) {
        float v = dm + acc[tm][tn][r];
        lsum += (v > 0.f) ? v : 0.f;
      }
    }
  }
#pragma unroll
  for (int off = 16; off > 0; off >>= 1) lsum += __shfl_xor(lsum, off, 32);
  __shared__ float s_part[8];
  if (lane == 0) s_part[waveId] = lsum;
  __syncthreads();
  if (tid == 0) {
    float t = 0.f;
#pragma unroll
    for (int j = 0; j < 8; ++j) t += s_part[j];
    partials[blockIdx.y * gridDim.x + blockIdx.x] = t;
  }
}

// ---------------- Pass 4: deterministic final reduction --------------------
__global__ void __launch_bounds__(256)
reduce_kernel(const float* __restrict__ partials, int n, float* __restrict__ out) {
  float s = 0.f;
  for (int i = threadIdx.x; i < n; i += 256) s += partials[i];
#pragma unroll
  for (int off = 16; off > 0; off >>= 1) s += __shfl_xor(s, off, 32);
  __shared__ float sm[8];
  const int w = threadIdx.x >> 5, lane = threadIdx.x & 31;
  if (lane == 0) sm[w] = s;
  __syncthreads();
  if (threadIdx.x == 0) {
    float t = 0.f;
#pragma unroll
    for (int j = 0; j < 8; ++j) t += sm[j];
    out[0] = t;
  }
}

extern "C" void kernel_launch(void* const* d_in, const int* in_sizes, int n_in,
                              void* d_out, int out_size, void* d_ws, size_t ws_size,
                              hipStream_t stream) {
  const float* q     = (const float*)d_in[0];
  const float* k     = (const float*)d_in[1];
  const float* queue = (const float*)d_in[2];

  const int B = in_sizes[0] / D_DIM;  // 4096
  const int N = in_sizes[2] / D_DIM;  // 8192

  char* ws = (char*)d_ws;
  _Float16* qh     = (_Float16*)ws;
  _Float16* queueh = (_Float16*)(ws + (size_t)B * D_DIM * 2);
  float*    pos    = (float*)(ws + (size_t)B * D_DIM * 2 + (size_t)N * D_DIM * 2);
  float*    parts  = (float*)((char*)pos + (size_t)B * 4);

  norm_qk_kernel<<<B, 256, 0, stream>>>(q, k, qh, pos);
  norm_queue_kernel<<<N, 256, 0, stream>>>(queue, queueh);

  dim3 grid(N / 128, B / 128);
  hinge_gemm_kernel<<<grid, 256, 0, stream>>>(qh, queueh, pos, parts);

  const int nPart = (N / 128) * (B / 128);
  reduce_kernel<<<1, 256, 0, stream>>>(parts, nPart, (float*)d_out);
}
